// BaseMPNN_87093346828879
// MI455X (gfx1250) — matrix operand order repaired and checked by reading, not compiled, not run on hardware
//
#include <hip/hip_runtime.h>
#include <hip/hip_bf16.h>

// ---------------------------------------------------------------------------
// MPNN forward for MI455X (gfx1250, wave32, WMMA 16x16x32 f16 -> f32 acc).
//
// Problem sizes (from reference): N=50000 nodes, E=800000 edges,
// F_X=64, F_E=32, H=128, 3 meta-layers.
//
// All large GEMMs (edge MLP, message MLP, node MLP) run through
// v_wmma_f32_16x16x32_f16. Block tile = 128 rows x 128 cols, 8 waves/block,
// each wave owns a 16x128 strip (8 f32 accumulators of v8f). A-tiles are
// gathered to LDS as f16; B (weights) are pre-packed into per-fragment
// lane-major f16 so each lane loads its v16h with one contiguous 32B read.
// Scatter-mean uses global_atomic_add_f32 into an L2-resident accumulator.
// ---------------------------------------------------------------------------

typedef __attribute__((ext_vector_type(16))) _Float16 v16h;
typedef __attribute__((ext_vector_type(8)))  _Float16 v8h;
typedef __attribute__((ext_vector_type(8)))  float    v8f;

#define NNODES 50000
#define NEDGES 800000
#define HDIM   128

// ---------------------------------------------------------------------------
// Pack weight W[out=128, in=K] (row-major, f32) into f16 WMMA B fragments.
// Fragment (kt,nt) covers K rows kt*32..+31, N cols nt*16..+15.
// Lane l (0..31): N = nt*16 + (l&15), K half = (l>>4)*16; 16 halves per lane.
// Storage: frag-major [kt*8+nt][lane][16] -> one 32B load per lane per frag.
// ---------------------------------------------------------------------------
__global__ void pack_w_kernel(const float* __restrict__ W,
                              _Float16* __restrict__ out, int K) {
  int tid = blockIdx.x * blockDim.x + threadIdx.x;
  if (tid >= K * 128) return;
  int e    = tid & 15;
  int lane = (tid >> 4) & 31;
  int frag = tid >> 9;
  int kt = frag >> 3;
  int nt = frag & 7;
  int k = kt * 32 + ((lane >> 4) << 4) + e;
  int n = nt * 16 + (lane & 15);
  float v = (k < K) ? W[(size_t)n * K + k] : 0.0f;
  out[tid] = (_Float16)v;
}

__global__ void zero_f32_kernel(float* __restrict__ p, long n) {
  long stride = (long)gridDim.x * blockDim.x;
  for (long i = (long)blockIdx.x * blockDim.x + threadIdx.x; i < n; i += stride)
    p[i] = 0.0f;
}

__global__ void degree_kernel(const int* __restrict__ col,
                              float* __restrict__ cnt, int E, int N) {
  int e = blockIdx.x * blockDim.x + threadIdx.x;
  if (e < E) {
    int c = col[e];
    c = c < 0 ? 0 : (c >= N ? N - 1 : c);
    atomicAdd(cnt + c, 1.0f);
  }
}

// ---------------------------------------------------------------------------
// Generic WMMA GEMM:   D[m, 0..127] = relu( A[m, 0..K-1] @ B[K,128] + bias )
// mode 0 (edge):  A row = [ X[row[m]] (xw) | X[col[m]] (xw) | Ein[m] (ew) ],
//                 store D to Out (E x 128).
// mode 1 (msg):   A row = [ X[row[m]] (xw) | Ein[m] (128) ],
//                 atomicAdd D into Out[col[m]*128 + n]  (scatter-sum).
// mode 2 (node):  A row = [ X[m] (xw) | S[m]/cnt[m] (128, guarded) ],
//                 store D to Out (N x 128), tail rows zero-staged/guarded.
// Block: 256 threads (8 waves), tile 128 rows x 128 cols.
// Dynamic LDS: 128*K f16 A-tile + 128 row idx + 128 col idx.
// ---------------------------------------------------------------------------
__global__ __launch_bounds__(256)
void mpnn_gemm_kernel(int mode,
                      const float* __restrict__ X, int xw,
                      const float* __restrict__ Ein, int ew,
                      const int* __restrict__ row, const int* __restrict__ col,
                      const float* __restrict__ cnt,
                      const _Float16* __restrict__ Bp,
                      const float* __restrict__ bias,
                      float* __restrict__ Out,
                      int M, int Ntot, int K) {
  extern __shared__ char smem[];
  _Float16* At = (_Float16*)smem;
  int* ridx = (int*)(smem + (size_t)128 * K * sizeof(_Float16));
  int* cidx = ridx + 128;

  const int tid   = threadIdx.x;
  const int mbase = blockIdx.x * 128;

  if (mode != 2) {
    for (int i = tid; i < 128; i += 256) {
      int m = mbase + i;                    // E is a multiple of 128: always valid
      int r = row[m], c = col[m];
      ridx[i] = r < 0 ? 0 : (r >= Ntot ? Ntot - 1 : r);
      cidx[i] = c < 0 ? 0 : (c >= Ntot ? Ntot - 1 : c);
    }
  }
  __syncthreads();

  // Stage A tile: 2 threads per row, each covers K/2 consecutive columns.
  {
    const int m    = tid >> 1;
    const int kbeg = (tid & 1) * (K >> 1);
    const int kend = kbeg + (K >> 1);
    for (int k = kbeg; k < kend; ++k) {
      float v = 0.0f;
      if (mode == 0) {
        if (k < xw)          v = X[(size_t)ridx[m] * xw + k];
        else if (k < 2 * xw) v = X[(size_t)cidx[m] * xw + (k - xw)];
        else                 v = Ein[(size_t)(mbase + m) * ew + (k - 2 * xw)];
      } else if (mode == 1) {
        if (k < xw)          v = X[(size_t)ridx[m] * xw + k];
        else                 v = Ein[(size_t)(mbase + m) * HDIM + (k - xw)];
      } else {
        int node = mbase + m;
        if (node < M) {
          if (k < xw) v = X[(size_t)node * xw + k];
          else {
            float c = cnt[node];
            v = (c > 0.0f)
                  ? Ein[(size_t)node * HDIM + (k - xw)] / fmaxf(c, 1.0f)
                  : 0.0f;
          }
        }
      }
      At[(size_t)m * K + k] = (_Float16)v;
    }
  }
  __syncthreads();

  const int lane  = tid & 31;
  const int wave  = tid >> 5;
  const int mrow  = wave * 16 + (lane & 15);   // A-fragment row (ISA 7.12.2)
  const int khalf = lane >> 4;

  v8f acc[8] = {};

  for (int k0 = 0; k0 < K; k0 += 32) {
    // A fragment: lanes 0-15 K[k0+0..15], lanes 16-31 K[k0+16..31].
    const _Float16* ap = At + (size_t)mrow * K + k0 + khalf * 16;
    union { v16h v; struct { v8h lo, hi; } p; } a;
    a.p.lo = *(const v8h*)(ap);
    a.p.hi = *(const v8h*)(ap + 8);

    const _Float16* bp = Bp + ((size_t)(k0 >> 5) * 8) * 512 + lane * 16;
#pragma unroll
    for (int nt = 0; nt < 8; ++nt) {
      v16h b = *(const v16h*)(bp + nt * 512);
      acc[nt] = __builtin_amdgcn_wmma_f32_16x16x32_f16(
          false, a.v, false, b, (short)0, acc[nt], false, false);
    }
  }

  // Epilogue. D layout: VGPR r, lanes 0-15 -> M=r, lanes 16-31 -> M=8+r;
  // N = (lane&15) within the 16-wide n-tile.
  const int nlo = lane & 15;
  const int mhi = (lane >> 4) * 8;
#pragma unroll
  for (int nt = 0; nt < 8; ++nt) {
    const int n  = nt * 16 + nlo;
    const float bn = bias[n];
#pragma unroll
    for (int r = 0; r < 8; ++r) {
      const int ml = wave * 16 + mhi + r;
      float v = acc[nt][r] + bn;
      v = v > 0.0f ? v : 0.0f;
      if (mode == 0) {
        Out[(size_t)(mbase + ml) * HDIM + n] = v;
      } else if (mode == 1) {
        atomicAdd(&Out[(size_t)cidx[ml] * HDIM + n], v);
      } else {
        int node = mbase + ml;
        if (node < M) Out[(size_t)node * HDIM + n] = v;
      }
    }
  }
}

// ---------------------------------------------------------------------------
// Column sum / sum-of-squares over a [rows x 128] matrix (for BN + readout).
// ---------------------------------------------------------------------------
__global__ __launch_bounds__(256)
void col_stats_kernel(const float* __restrict__ buf, long rows,
                      float* __restrict__ sum, float* __restrict__ sumsq) {
  const int tid  = threadIdx.x;
  const int colv = tid & 127;
  const int half = tid >> 7;
  float s = 0.0f, q = 0.0f;
  for (long r = (long)blockIdx.x * 2 + half; r < rows; r += (long)gridDim.x * 2) {
    float v = buf[r * HDIM + colv];
    s += v; q += v * v;
  }
  __shared__ float ls[256], lq[256];
  ls[tid] = s; lq[tid] = q;
  __syncthreads();
  if (tid < 128) {
    atomicAdd(&sum[tid],   ls[tid] + ls[tid + 128]);
    atomicAdd(&sumsq[tid], lq[tid] + lq[tid + 128]);
  }
}

__global__ void bn_apply_kernel(float* __restrict__ buf, long rows,
                                const float* __restrict__ sum,
                                const float* __restrict__ sumsq,
                                const float* __restrict__ g,
                                const float* __restrict__ b) {
  const long total  = rows * HDIM;
  const long stride = (long)gridDim.x * blockDim.x;
  const float inv_r = 1.0f / (float)rows;
  for (long idx = (long)blockIdx.x * blockDim.x + threadIdx.x; idx < total;
       idx += stride) {
    int colv  = (int)(idx & 127);
    float mu  = sum[colv] * inv_r;
    float var = sumsq[colv] * inv_r - mu * mu;
    float v   = buf[idx];
    buf[idx]  = (v - mu) * rsqrtf(var + 1e-5f) * g[colv] + b[colv];
  }
}

__global__ void readout_kernel(const float* __restrict__ nsum,
                               const float* __restrict__ esum,
                               const float* __restrict__ regW,
                               const float* __restrict__ regb,
                               float* __restrict__ out,
                               float invN, float invE) {
  if (blockIdx.x == 0 && threadIdx.x == 0) {
    float acc = regb[0];
    for (int h = 0; h < HDIM; ++h) acc += nsum[h] * invN * regW[h];
    for (int h = 0; h < HDIM; ++h) acc += esum[h] * invE * regW[HDIM + h];
    out[0] = acc;
  }
}

// ---------------------------------------------------------------------------
// Host orchestration.
// d_in flatten order (setup_inputs dict order, params dict flattened):
//   0: x[50000,64]  1: edge_index[2,800000] (int)  2: edge_attr[800000,32]
//   3*: per layer i (base 3+6i): eW, eb, nW1, nb1, nW2, nb2
//   21: bn_node_g 22: bn_node_b 23: bn_edge_g 24: bn_edge_b 25: regW 26: regb
// (layer weights occupy indices 3..20)
// ---------------------------------------------------------------------------
extern "C" void kernel_launch(void* const* d_in, const int* in_sizes, int n_in,
                              void* d_out, int out_size, void* d_ws, size_t ws_size,
                              hipStream_t stream) {
  (void)in_sizes; (void)n_in; (void)out_size; (void)ws_size;
  const int N = NNODES, E = NEDGES;

  const float* x0  = (const float*)d_in[0];
  const int*   ei  = (const int*)d_in[1];
  const int*   row = ei;
  const int*   col = ei + E;
  const float* ea0 = (const float*)d_in[2];

  const float* eW[3];  const float* eb[3];
  const float* nW1[3]; const float* nb1[3];
  const float* nW2[3]; const float* nb2[3];
  for (int i = 0; i < 3; ++i) {
    int base = 3 + 6 * i;
    eW[i]  = (const float*)d_in[base + 0];
    eb[i]  = (const float*)d_in[base + 1];
    nW1[i] = (const float*)d_in[base + 2];
    nb1[i] = (const float*)d_in[base + 3];
    nW2[i] = (const float*)d_in[base + 4];
    nb2[i] = (const float*)d_in[base + 5];
  }
  const float* bn_ng = (const float*)d_in[21];
  const float* bn_nb = (const float*)d_in[22];
  const float* bn_eg = (const float*)d_in[23];
  const float* bn_eb = (const float*)d_in[24];
  const float* regW  = (const float*)d_in[25];
  const float* regb  = (const float*)d_in[26];

  // Workspace carve-up (~896 MB).
  char* ws = (char*)d_ws;
  size_t off = 0;
  auto carve = [&](size_t bytes) {
    size_t o = off;
    off = (off + bytes + 255) & ~(size_t)255;
    return o;
  };
  const size_t ebytes = (size_t)E * HDIM * sizeof(float);
  const size_t xbytes = (size_t)N * HDIM * sizeof(float);
  float* ebuf0 = (float*)(ws + carve(ebytes));
  float* ebuf1 = (float*)(ws + carve(ebytes));
  float* xbuf0 = (float*)(ws + carve(xbytes));
  float* xbuf1 = (float*)(ws + carve(xbytes));
  float* sbuf  = (float*)(ws + carve(xbytes));
  float* cnt   = (float*)(ws + carve((size_t)N * sizeof(float)));
  float* stats = (float*)(ws + carve(512 * sizeof(float))); // [sum|sq|sum2|sq2]

  const int Ke[3]  = {2 * 64 + 32, 2 * HDIM + HDIM, 2 * HDIM + HDIM}; // 160,384,384
  const int Kn[3]  = {64 + HDIM, HDIM + HDIM, HDIM + HDIM};           // 192,256,256
  _Float16* peW[3]; _Float16* pnW1[3]; _Float16* pnW2[3];
  for (int i = 0; i < 3; ++i) {
    peW[i]  = (_Float16*)(ws + carve((size_t)Ke[i] * HDIM * sizeof(_Float16)));
    pnW1[i] = (_Float16*)(ws + carve((size_t)Kn[i] * HDIM * sizeof(_Float16)));
    pnW2[i] = (_Float16*)(ws + carve((size_t)Kn[i] * HDIM * sizeof(_Float16)));
  }

  // --- weight packing (f32 -> f16 WMMA fragments) ---
  for (int i = 0; i < 3; ++i) {
    pack_w_kernel<<<(Ke[i] * 128 + 255) / 256, 256, 0, stream>>>(eW[i],  peW[i],  Ke[i]);
    pack_w_kernel<<<(Kn[i] * 128 + 255) / 256, 256, 0, stream>>>(nW1[i], pnW1[i], Kn[i]);
    pack_w_kernel<<<(Kn[i] * 128 + 255) / 256, 256, 0, stream>>>(nW2[i], pnW2[i], Kn[i]);
  }

  // --- in-degree (layer-invariant) ---
  zero_f32_kernel<<<256, 256, 0, stream>>>(cnt, (long)N);
  degree_kernel<<<(E + 255) / 256, 256, 0, stream>>>(col, cnt, E, N);

  const dim3 egrid(E / 128);                  // 6250, exact
  const dim3 ngrid((N + 127) / 128);          // 391
  auto shmem = [](int K) { return (size_t)128 * K * sizeof(_Float16) + 256 * sizeof(int); };

  const float* xin  = x0;   int xw = 64;
  const float* ein  = ea0;  int ew = 32;
  float* eping[3] = {ebuf0, ebuf1, ebuf0};
  float* xping[3] = {xbuf0, xbuf1, xbuf0};

  for (int L = 0; L < 3; ++L) {
    float* eout = eping[L];
    float* xout = xping[L];

    // EdgeModel: eout = relu([x[row]|x[col]|ein] @ eW^T + eb)
    mpnn_gemm_kernel<<<egrid, 256, shmem(Ke[L]), stream>>>(
        0, xin, xw, ein, ew, row, col, nullptr, peW[L], eb[L], eout, E, N, Ke[L]);

    // NodeModel: msg GEMM + scatter-sum into sbuf
    zero_f32_kernel<<<4096, 256, 0, stream>>>(sbuf, (long)N * HDIM);
    mpnn_gemm_kernel<<<egrid, 256, shmem(Kn[L]), stream>>>(
        1, xin, xw, eout, HDIM, row, col, nullptr, pnW1[L], nb1[L], sbuf, E, N, Kn[L]);

    // xout = relu([x | sbuf/cnt] @ nW2^T + nb2)
    mpnn_gemm_kernel<<<ngrid, 256, shmem(Kn[L]), stream>>>(
        2, xin, xw, sbuf, HDIM, nullptr, nullptr, cnt, pnW2[L], nb2[L], xout, N, N, Kn[L]);

    if (L < 2) {
      // BatchNorm (training-mode batch stats) on nodes then edges, in place.
      zero_f32_kernel<<<2, 256, 0, stream>>>(stats, 256);
      col_stats_kernel<<<1024, 256, 0, stream>>>(xout, (long)N, stats, stats + 128);
      bn_apply_kernel<<<8192, 256, 0, stream>>>(xout, (long)N, stats, stats + 128,
                                                bn_ng, bn_nb);
      zero_f32_kernel<<<2, 256, 0, stream>>>(stats, 256);
      col_stats_kernel<<<2048, 256, 0, stream>>>(eout, (long)E, stats, stats + 128);
      bn_apply_kernel<<<16384, 256, 0, stream>>>(eout, (long)E, stats, stats + 128,
                                                 bn_eg, bn_eb);
    }
    xin = xout; xw = HDIM;
    ein = eout; ew = HDIM;
  }

  // Readout: column means of final x (xbuf0) and edge_attr (ebuf0), dot regW.
  zero_f32_kernel<<<4, 256, 0, stream>>>(stats, 512);
  col_stats_kernel<<<1024, 256, 0, stream>>>(xping[2], (long)N, stats,       stats + 128);
  col_stats_kernel<<<2048, 256, 0, stream>>>(eping[2], (long)E, stats + 256, stats + 384);
  readout_kernel<<<1, 64, 0, stream>>>(stats, stats + 256, regW, regb,
                                       (float*)d_out, 1.0f / (float)N, 1.0f / (float)E);
}
